// CARM_90185723281482
// MI455X (gfx1250) — matrix operand order repaired and checked by reading, not compiled, not run on hardware
//
#include <hip/hip_runtime.h>
#include <hip/hip_bf16.h>

typedef __attribute__((ext_vector_type(16))) __bf16 v16bf;
typedef __attribute__((ext_vector_type(8)))  float  v8f;
typedef __attribute__((ext_vector_type(2)))  float  v2f;

#define GCN_C 64
#define GCN_F 512
#define GCN_D 64
#define BN_EPS 1e-3f

// ---------------------------------------------------------------------------
// Prologue 1: normalized adjacency  adjn = D^{-1/2} A D^{-1/2} + I   [64x64]
// ---------------------------------------------------------------------------
__global__ __launch_bounds__(64) void gcn_adj_kernel(
    const float* __restrict__ A, float* __restrict__ adjn)
{
    __shared__ float dinv[GCN_C];
    const int i = threadIdx.x;                 // 64 threads, one row each
    float s = 0.0f;
    #pragma unroll 8
    for (int j = 0; j < GCN_C; ++j) s += fabsf(A[i * GCN_C + j]);
    s = fmaxf(s, 1e-8f);
    dinv[i] = rsqrtf(s);
    __syncthreads();
    const float di = dinv[i];
    #pragma unroll 8
    for (int j = 0; j < GCN_C; ++j) {
        float v = A[i * GCN_C + j] * di * dinv[j];
        if (i == j) v += 1.0f;
        adjn[i * GCN_C + j] = v;
    }
}

// ---------------------------------------------------------------------------
// Prologue 2: pack Theta [F=512, D=64] fp32 -> bf16 in WMMA B-fragment order.
// Wp layout: [ntile 0..3][kstep 0..15][lane 0..31][j 0..15]  (32768 bf16)
//   element (K, N) with K = kstep*32 + (j<8 ? 8*(lane/16)+j : 16+8*(lane/16)+j-8), N = ntile*16 + lane%16
// so one wave's B fragment for (kstep, ntile) is a contiguous 32B chunk/lane.
// ---------------------------------------------------------------------------
__global__ __launch_bounds__(256) void gcn_pack_kernel(
    const float* __restrict__ W, __bf16* __restrict__ Wp)
{
    const int t = blockIdx.x * blockDim.x + threadIdx.x;
    if (t >= 4 * 16 * 32 * 16) return;
    const int j    = t & 15;
    const int lane = (t >> 4) & 31;
    const int ks   = (t >> 9) & 15;
    const int nt   = t >> 13;
    const int K = ks * 32 + ((j < 8) ? (8 * (lane >> 4) + j) : (16 + 8 * (lane >> 4) + (j - 8)));
    const int N = nt * 16 + (lane & 15);
    Wp[t] = (__bf16)W[K * GCN_D + N];
}

// ---------------------------------------------------------------------------
// Main kernel: one block (4 wave32) per batch element b.
//   Stage packed Theta (64 KB bf16) -> LDS via GLOBAL_LOAD_ASYNC_TO_LDS_B128
//   S = X_b[64x512] @ Theta[512x64]   (bf16 WMMA, f32 accum, B from LDS)
//   O = adjn[64x64] @ S               (f32 WMMA 16x16x4, S staged in LDS)
//   out = BN(ELU(O + bias))
// Wave w owns rows [16w, 16w+16); 4 accumulators cover all 64 columns.
// ---------------------------------------------------------------------------
__global__ __launch_bounds__(128) void gcn_main_kernel(
    const float*  __restrict__ x,      // [B,64,512]
    const float*  __restrict__ adjn,   // [64,64]      (ws)
    const __bf16* __restrict__ Wp,     // packed Theta (ws)
    const float*  __restrict__ bias,
    const float*  __restrict__ gamma,
    const float*  __restrict__ beta,
    const float*  __restrict__ mmean,
    const float*  __restrict__ mvar,
    float*        __restrict__ out)    // [B,64,64]
{
    constexpr int SSTR = 68;                       // padded LDS row stride
    // Theta stage (64 KB) and S stage (17 KB) are not live at the same time.
    __shared__ union {
        __bf16 w[4 * 16 * 32 * 16];                // 65536 B
        float  s[GCN_C * SSTR];                    // 17408 B
    } lds;

    const int b     = blockIdx.x;
    const int lane  = threadIdx.x & 31;
    const int wid   = threadIdx.x >> 5;            // 0..3
    const int half  = lane >> 4;                   // 0/1
    const int l16   = lane & 15;
    const int mbase = wid * 16;

    // ---------------- Async-stage packed Theta into LDS ---------------------
    // Each wave copies its 16 KB quarter: 32 x (32 lanes x 16 B) transfers.
    {
        const unsigned ldsW = (unsigned)(size_t)(void*)&lds.w[0];
        const char*    src  = (const char*)Wp + wid * 16384 + lane * 16;
        const unsigned dst  = ldsW + (unsigned)(wid * 16384 + lane * 16);
        #pragma unroll
        for (int i = 0; i < 32; ++i) {
            asm volatile("global_load_async_to_lds_b128 %0, %1, off"
                         :: "v"(dst + (unsigned)(i * 512)),
                            "v"(src + (size_t)i * 512)
                         : "memory");
        }
        asm volatile("s_wait_asynccnt 0x0" ::: "memory");
    }
    __syncthreads();

    // ---------------- GEMM 1: S = X_b @ Theta  (K = 512, 16 steps of 32) ----
    v8f acc0 = {}, acc1 = {}, acc2 = {}, acc3 = {};

    // A fragment source: row (mbase + l16); per k-step two 8-float runs:
    //   run0 = Kb + 8*half + [0..7],  run1 = Kb + 16 + 8*half + [0..7]
    const float* xrow = x + ((size_t)b * GCN_C + (mbase + l16)) * GCN_F
                          + 8 * half;

    #pragma unroll 4
    for (int ks = 0; ks < 16; ++ks) {
        const float* p = xrow + ks * 32;
        const float4 q0 = *(const float4*)(p +  0);
        const float4 q1 = *(const float4*)(p +  4);
        const float4 q2 = *(const float4*)(p + 16);
        const float4 q3 = *(const float4*)(p + 20);
        v16bf a;
        a[0]  = (__bf16)q0.x; a[1]  = (__bf16)q0.y;
        a[2]  = (__bf16)q0.z; a[3]  = (__bf16)q0.w;
        a[4]  = (__bf16)q1.x; a[5]  = (__bf16)q1.y;
        a[6]  = (__bf16)q1.z; a[7]  = (__bf16)q1.w;
        a[8]  = (__bf16)q2.x; a[9]  = (__bf16)q2.y;
        a[10] = (__bf16)q2.z; a[11] = (__bf16)q2.w;
        a[12] = (__bf16)q3.x; a[13] = (__bf16)q3.y;
        a[14] = (__bf16)q3.z; a[15] = (__bf16)q3.w;

        // B fragments from LDS: offset = nt*8192 + ks*512 + lane*16 (bf16 elems)
        const __bf16* wb = lds.w + ((size_t)ks * 32 + lane) * 16;
        const v16bf b0 = *(const v16bf*)(wb + 0 * 8192);
        const v16bf b1 = *(const v16bf*)(wb + 1 * 8192);
        const v16bf b2 = *(const v16bf*)(wb + 2 * 8192);
        const v16bf b3 = *(const v16bf*)(wb + 3 * 8192);

        acc0 = __builtin_amdgcn_wmma_f32_16x16x32_bf16(false, a, false, b0,
                                                       (short)0, acc0, false, false);
        acc1 = __builtin_amdgcn_wmma_f32_16x16x32_bf16(false, a, false, b1,
                                                       (short)0, acc1, false, false);
        acc2 = __builtin_amdgcn_wmma_f32_16x16x32_bf16(false, a, false, b2,
                                                       (short)0, acc2, false, false);
        acc3 = __builtin_amdgcn_wmma_f32_16x16x32_bf16(false, a, false, b3,
                                                       (short)0, acc3, false, false);
    }

    // All Theta reads done before S overwrites the same LDS bytes.
    __syncthreads();

    // Stage S in LDS: C/D layout -> element (M = mbase + r + 8*half, N)
    {
        v8f* accs[4] = { &acc0, &acc1, &acc2, &acc3 };
        #pragma unroll
        for (int nt = 0; nt < 4; ++nt) {
            const int N = nt * 16 + l16;
            #pragma unroll
            for (int r = 0; r < 8; ++r) {
                const int M = mbase + r + 8 * half;
                lds.s[M * SSTR + N] = (*accs[nt])[r];
            }
        }
    }
    __syncthreads();

    // ---------------- GEMM 2: O = adjn @ S  (f32 WMMA, K = 64, steps of 4) --
    v8f o0 = {}, o1 = {}, o2 = {}, o3 = {};
    const float* arow = adjn + (mbase + l16) * GCN_C + 2 * half;

    #pragma unroll 4
    for (int k = 0; k < GCN_C; k += 4) {
        const v2f af = *(const v2f*)(arow + k);      // A: K = k + 2*half + {0,1}
        const int kr = (k + 2 * half) * SSTR;
        #pragma unroll
        for (int nt = 0; nt < 4; ++nt) {
            const int N = nt * 16 + l16;
            v2f bf;
            bf[0] = lds.s[kr + N];
            bf[1] = lds.s[kr + SSTR + N];
            v8f* oacc = (nt == 0) ? &o0 : (nt == 1) ? &o1 : (nt == 2) ? &o2 : &o3;
            *oacc = __builtin_amdgcn_wmma_f32_16x16x4_f32(false, af, false, bf,
                                                          (short)0, *oacc, false, false);
        }
    }

    // ---------------- Epilogue: bias + ELU + inference BatchNorm ------------
    {
        v8f* oaccs[4] = { &o0, &o1, &o2, &o3 };
        float* outb = out + (size_t)b * GCN_C * GCN_D;
        #pragma unroll
        for (int nt = 0; nt < 4; ++nt) {
            const int N  = nt * 16 + l16;
            const float bs = bias[N];
            const float gm = gamma[N];
            const float bt = beta[N];
            const float mn = mmean[N];
            const float is = rsqrtf(mvar[N] + BN_EPS);
            #pragma unroll
            for (int r = 0; r < 8; ++r) {
                const int M = mbase + r + 8 * half;
                float v = (*oaccs[nt])[r] + bs;
                v = (v > 0.0f) ? v : expm1f(v);        // ELU, alpha = 1
                v = (v - mn) * is * gm + bt;
                outb[M * GCN_D + N] = v;
            }
        }
    }
}

// ---------------------------------------------------------------------------
extern "C" void kernel_launch(void* const* d_in, const int* in_sizes, int n_in,
                              void* d_out, int out_size, void* d_ws, size_t ws_size,
                              hipStream_t stream)
{
    const float* x     = (const float*)d_in[0];  // [B,64,512]
    const float* adjw  = (const float*)d_in[1];  // [64,64]
    const float* W     = (const float*)d_in[2];  // [512,64]
    const float* bias  = (const float*)d_in[3];
    const float* gamma = (const float*)d_in[4];
    const float* beta  = (const float*)d_in[5];
    const float* mmean = (const float*)d_in[6];
    const float* mvar  = (const float*)d_in[7];
    float* out = (float*)d_out;

    const int B = in_sizes[0] / (GCN_C * GCN_F);

    float*  adjn = (float*)d_ws;                                    // 16 KB
    __bf16* Wp   = (__bf16*)((char*)d_ws + GCN_C * GCN_C * sizeof(float)); // 64 KB

    gcn_adj_kernel <<<1,   64,  0, stream>>>(adjw, adjn);
    gcn_pack_kernel<<<128, 256, 0, stream>>>(W, Wp);
    gcn_main_kernel<<<B,   128, 0, stream>>>(x, adjn, Wp, bias, gamma, beta,
                                             mmean, mvar, out);
}
